// GraphNeuralPPOPolicy_21749714387569
// MI455X (gfx1250) — matrix operand order repaired and checked by reading, not compiled, not run on hardware
//
#include <hip/hip_runtime.h>
#include <hip/hip_bf16.h>

#define NNODES 10000
#define NEDGES 320000
#define INF_ 256
#define OUTF 32
#define NHEADS 8
#define HOF 256   // NHEADS*OUTF

typedef __attribute__((ext_vector_type(16))) _Float16 v16h;
typedef __attribute__((ext_vector_type(8)))  float    v8f;

// order-preserving float<->uint encoding so atomicMax(uint) == float max
__device__ __forceinline__ unsigned fenc(float f) {
    unsigned u = __float_as_uint(f);
    return (u & 0x80000000u) ? ~u : (u | 0x80000000u);
}
__device__ __forceinline__ float fdec(unsigned u) {
    unsigned b = (u & 0x80000000u) ? (u ^ 0x80000000u) : ~u;
    return __uint_as_float(b);
}
#define ENC_NEG_INF 0x007FFFFFu   // fenc(-inf)

// ---------------- init: out = bias, nodemax = -inf, nodesum = 0 ----------------
__global__ void k_init(float* __restrict__ out, const float* __restrict__ bias,
                       unsigned* __restrict__ nodemax, float* __restrict__ nodesum) {
    int i = blockIdx.x * blockDim.x + threadIdx.x;
    if (i < NNODES * HOF) out[i] = bias[i & (HOF - 1)];
    if (i < NNODES * NHEADS) { nodemax[i] = ENC_NEG_INF; nodesum[i] = 0.0f; }
}

// ---------------- convert x -> f16, W -> f16 transposed ----------------
__global__ void k_convert(const float* __restrict__ x, const float* __restrict__ W,
                          _Float16* __restrict__ xh, _Float16* __restrict__ WhT) {
    int i = blockIdx.x * blockDim.x + threadIdx.x;
    if (i < NNODES * INF_) xh[i] = (_Float16)x[i];
    if (i < HOF * INF_) {
        int n = i >> 8, k = i & 255;            // WhT[n][k] = W[k][n]
        WhT[i] = (_Float16)W[k * HOF + n];
    }
}

// ---------------- h = x @ W via v_wmma_f32_16x16x32_f16 ----------------
// one wave per 16x16 output tile; K loop of 8 WMMAs (K=256)
__global__ __launch_bounds__(256) void k_gemm(const _Float16* __restrict__ xh,
                                              const _Float16* __restrict__ WhT,
                                              float* __restrict__ hbuf) {
    int wave = threadIdx.x >> 5;
    int lane = threadIdx.x & 31;
    int tile = blockIdx.x * 8 + wave;           // 625*16 = 10000 tiles exactly
    if (tile >= (NNODES / 16) * 16) return;
    int mt = tile >> 4, nt = tile & 15;
    int m0 = mt * 16, n0 = nt * 16;
    int lr = lane & 15, hh = lane >> 4;

    // ISA 7.12.2: 16-bit A 16x32 — lane M=lr, halves of K per lane group
    const unsigned* arow = (const unsigned*)(xh + (size_t)(m0 + lr) * INF_);
    // B 32x16 — lane N=lr; WhT row-per-N makes K pairs contiguous
    const unsigned* brow = (const unsigned*)(WhT + (size_t)(n0 + lr) * INF_);

    v8f acc = {};
    for (int kb = 0; kb < INF_ / 32; ++kb) {
        union { v16h v; unsigned u[8]; } A, B;
#pragma unroll
        for (int v = 0; v < 8; ++v) {
            int Ka = kb * 32 + ((v >> 2) << 4) + (hh << 3) + ((v & 3) << 1);
            A.u[v] = arow[Ka >> 1];
            int Kb = kb * 32 + ((v >> 2) << 4) + (hh << 3) + ((v & 3) << 1);
            B.u[v] = brow[Kb >> 1];
        }
        acc = __builtin_amdgcn_wmma_f32_16x16x32_f16(false, A.v, false, B.v,
                                                     (short)0, acc, false, false);
    }
    // C/D layout: VGPR r -> M = r + 8*hh, N = lr
#pragma unroll
    for (int r = 0; r < 8; ++r)
        hbuf[(size_t)(m0 + r + (hh << 3)) * HOF + n0 + lr] = acc[r];
}

// ---------------- per-node attention scores ----------------
__global__ void k_scores(const float* __restrict__ hbuf, const float* __restrict__ a_src,
                         const float* __restrict__ a_dst,
                         float* __restrict__ s_src, float* __restrict__ s_dst) {
    int t = blockIdx.x * blockDim.x + threadIdx.x;
    if (t >= NNODES * NHEADS) return;
    int n = t >> 3, hd = t & 7;
    const float* hp = hbuf + (size_t)n * HOF + hd * OUTF;
    const float* as = a_src + hd * OUTF;
    const float* ad = a_dst + hd * OUTF;
    float ss = 0.f, sd = 0.f;
#pragma unroll
    for (int f = 0; f < OUTF; ++f) { float v = hp[f]; ss += v * as[f]; sd += v * ad[f]; }
    s_src[t] = ss; s_dst[t] = sd;
}

// ---------------- per-edge alpha + fused segment max ----------------
// warp per edge; lane = feature within head; 8 heads
__global__ __launch_bounds__(256) void k_alpha(const int* __restrict__ ei,
                                               const float* __restrict__ ea,
                                               const float* __restrict__ We,
                                               const float* __restrict__ be,
                                               const float* __restrict__ hbuf,
                                               const float* __restrict__ s_src,
                                               const float* __restrict__ s_dst,
                                               float* __restrict__ alpha,
                                               unsigned* __restrict__ nodemax) {
    int wid = (blockIdx.x * blockDim.x + threadIdx.x) >> 5;
    int lane = threadIdx.x & 31;
    if (wid >= NEDGES) return;
    int src = ei[wid], dst = ei[NEDGES + wid];
    float e0 = ea[wid * 3 + 0], e1 = ea[wid * 3 + 1], e2 = ea[wid * 3 + 2];
    const float* hd = hbuf + (size_t)dst * HOF;

    float part[NHEADS];
#pragma unroll
    for (int k = 0; k < NHEADS; ++k) {
        int f = k * OUTF + lane;
        float enc = fmaf(e0, We[f], fmaf(e1, We[HOF + f], fmaf(e2, We[2 * HOF + f], be[f])));
        enc = fmaxf(enc, 0.0f);                 // relu
        part[k] = enc * hd[f];
    }
#pragma unroll
    for (int off = 16; off > 0; off >>= 1)
#pragma unroll
        for (int k = 0; k < NHEADS; ++k)
            part[k] += __shfl_xor(part[k], off, 32);

    if (lane == 0) {
#pragma unroll
        for (int k = 0; k < NHEADS; ++k) {
            float a = s_src[src * NHEADS + k] + s_dst[dst * NHEADS + k] + part[k];
            a = (a >= 0.0f) ? a : 0.2f * a;     // leaky relu
            alpha[(size_t)wid * NHEADS + k] = a;
            atomicMax(&nodemax[dst * NHEADS + k], fenc(a));
        }
    }
}

// ---------------- exp(alpha - max) + segment sum ----------------
__global__ void k_expsum(const int* __restrict__ ei, float* __restrict__ alpha,
                         const unsigned* __restrict__ nodemax, float* __restrict__ nodesum) {
    int t = blockIdx.x * blockDim.x + threadIdx.x;
    if (t >= NEDGES * NHEADS) return;
    int e = t >> 3, hd = t & 7;
    int dst = ei[NEDGES + e];
    float m = fdec(nodemax[dst * NHEADS + hd]);
    float ex = __expf(alpha[t] - m);
    alpha[t] = ex;
    atomicAdd(&nodesum[dst * NHEADS + hd], ex);
}

// ---------------- message aggregation (scatter-add) ----------------
__global__ __launch_bounds__(256) void k_agg(const int* __restrict__ ei,
                                             const float* __restrict__ alpha,
                                             const float* __restrict__ nodesum,
                                             const float* __restrict__ hbuf,
                                             float* __restrict__ out) {
    int wid = (blockIdx.x * blockDim.x + threadIdx.x) >> 5;
    int lane = threadIdx.x & 31;
    if (wid >= NEDGES) return;
    int src = ei[wid], dst = ei[NEDGES + wid];
    const float* hs = hbuf + (size_t)src * HOF;
#pragma unroll
    for (int hd = 0; hd < NHEADS; ++hd) {
        float coef = alpha[(size_t)wid * NHEADS + hd] /
                     fmaxf(nodesum[dst * NHEADS + hd], 1e-10f);
        atomicAdd(&out[(size_t)dst * HOF + hd * OUTF + lane],
                  hs[hd * OUTF + lane] * coef);
    }
}

extern "C" void kernel_launch(void* const* d_in, const int* in_sizes, int n_in,
                              void* d_out, int out_size, void* d_ws, size_t ws_size,
                              hipStream_t stream) {
    (void)in_sizes; (void)n_in; (void)out_size; (void)ws_size;
    const float* x     = (const float*)d_in[0];
    const int*   ei    = (const int*)  d_in[1];
    const float* ea    = (const float*)d_in[2];
    const float* W     = (const float*)d_in[3];
    const float* a_src = (const float*)d_in[4];
    const float* a_dst = (const float*)d_in[5];
    const float* We    = (const float*)d_in[6];
    const float* be    = (const float*)d_in[7];
    const float* bias  = (const float*)d_in[8];
    float* out = (float*)d_out;

    char* w = (char*)d_ws;
    size_t o = 0;
    auto aln = [](size_t s) { return (s + 255) & ~(size_t)255; };
    _Float16* xh      = (_Float16*)(w + o); o += aln((size_t)NNODES * INF_ * 2);
    _Float16* WhT     = (_Float16*)(w + o); o += aln((size_t)HOF * INF_ * 2);
    float*    hbuf    = (float*)   (w + o); o += aln((size_t)NNODES * HOF * 4);
    float*    s_src   = (float*)   (w + o); o += aln((size_t)NNODES * NHEADS * 4);
    float*    s_dst   = (float*)   (w + o); o += aln((size_t)NNODES * NHEADS * 4);
    float*    alpha   = (float*)   (w + o); o += aln((size_t)NEDGES * NHEADS * 4);
    unsigned* nodemax = (unsigned*)(w + o); o += aln((size_t)NNODES * NHEADS * 4);
    float*    nodesum = (float*)   (w + o); o += aln((size_t)NNODES * NHEADS * 4);

    k_init   <<<(NNODES * HOF + 255) / 256, 256, 0, stream>>>(out, bias, nodemax, nodesum);
    k_convert<<<(NNODES * INF_ + 255) / 256, 256, 0, stream>>>(x, W, xh, WhT);
    k_gemm   <<<(NNODES / 16) * 16 / 8, 256, 0, stream>>>(xh, WhT, hbuf);
    k_scores <<<(NNODES * NHEADS + 255) / 256, 256, 0, stream>>>(hbuf, a_src, a_dst, s_src, s_dst);
    k_alpha  <<<NEDGES * 32 / 256, 256, 0, stream>>>(ei, ea, We, be, hbuf, s_src, s_dst, alpha, nodemax);
    k_expsum <<<NEDGES * NHEADS / 256, 256, 0, stream>>>(ei, alpha, nodemax, nodesum);
    k_agg    <<<NEDGES * 32 / 256, 256, 0, stream>>>(ei, alpha, nodesum, hbuf, out);
}
